// RWKV_2911987826817
// MI455X (gfx1250) — compile-verified, matrix operand order checked by reading, not run on hardware
//
#include <hip/hip_runtime.h>
#include <hip/hip_bf16.h>
#include <hip/hip_fp16.h>

typedef __attribute__((ext_vector_type(16))) _Float16 v16h;
typedef __attribute__((ext_vector_type(8)))  float    v8f;

#define B_      8
#define T_      1024
#define C_      512
#define H_      8
#define N_      64
#define M_      (B_ * T_)        /* 8192 tokens */
#define DLORA_  32
#define DGATE_  96
#define GN_EPS_ 6.4e-4f          /* 1e-5 * 8^2 */

// ---------------------------------------------------------------------------
// Depthwise 3x3 SAME conv on x viewed as (B, C, 32, 32); xx = conv(x) - x.
// x layout: [b][t][c], t = h*32+w.
// ---------------------------------------------------------------------------
__global__ __launch_bounds__(256) void conv_xx_kernel(const float* __restrict__ x,
                                                      const float* __restrict__ tsw,
                                                      float* __restrict__ xx) {
    int idx = blockIdx.x * 256 + threadIdx.x;          // over B*T*C
    int c = idx & (C_ - 1);
    int t = (idx >> 9) & (T_ - 1);
    int b = idx >> 19;
    int h = t >> 5, w = t & 31;
    const float* wp = tsw + c * 9;
    float acc = 0.f;
#pragma unroll
    for (int dh = -1; dh <= 1; ++dh) {
        int hh = h + dh;
        if (hh < 0 || hh >= 32) continue;
#pragma unroll
        for (int dw = -1; dw <= 1; ++dw) {
            int ww = w + dw;
            if (ww < 0 || ww >= 32) continue;
            acc += x[((size_t)b * T_ + (hh * 32 + ww)) * C_ + c] * wp[(dh + 1) * 3 + (dw + 1)];
        }
    }
    xx[idx] = acc - x[idx];
}

// ---------------------------------------------------------------------------
// Weight packing: fp32 -> f16, optionally transposed to [N][K] row-major.
// ---------------------------------------------------------------------------
__global__ __launch_bounds__(256) void cvt_f16_kernel(const float* __restrict__ src,
                                                      _Float16* __restrict__ dst, int n) {
    int i = blockIdx.x * 256 + threadIdx.x;
    if (i < n) dst[i] = (_Float16)src[i];
}

__global__ __launch_bounds__(256) void transpose_f16_kernel(const float* __restrict__ src,
                                                            _Float16* __restrict__ dst,
                                                            int K, int N) {
    int i = blockIdx.x * 256 + threadIdx.x;
    if (i < K * N) {
        int k = i / N, n = i - k * N;
        dst[(size_t)n * K + k] = (_Float16)src[i];
    }
}

// ---------------------------------------------------------------------------
// WMMA GEMM: Cout[m,n] = epilogue( sum_k A[m,k] * BT[n,k] )
// A is fp32 (optionally mixed: A + Axx*mix per channel k) converted to f16.
// BT is f16 [N][K] row-major.  One wave computes a 16(M) x 32(N) tile with
// two v_wmma_f32_16x16x32_f16 per 32-wide k step.  4 waves/block stack in M.
// ---------------------------------------------------------------------------
enum { EPI_NONE = 0, EPI_TANH, EPI_SIG, EPI_DEC, EPI_SIGB, EPI_VMIX };

__device__ __forceinline__ float sigmoidf_(float x) { return 1.f / (1.f + expf(-x)); }

template <int EPI>
__device__ __forceinline__ float epilogue_(float acc, int n, size_t off,
                                           const float* bias, const float* aux0,
                                           const float* aux1) {
    if (EPI == EPI_TANH) return tanhf(acc);
    if (EPI == EPI_SIG)  return sigmoidf_(acc);
    if (EPI == EPI_DEC) {
        float t = bias[n] + acc;
        float w = -logf(1.f + expf(-t)) - 0.5f;   // -softplus(-t) - 0.5
        return expf(-expf(w));                    // dec = exp(-exp(w))
    }
    if (EPI == EPI_SIGB) return sigmoidf_(bias[n] + acc);
    if (EPI == EPI_VMIX) {
        float s  = sigmoidf_(bias[n] + acc);
        float vr = aux0[off];
        return vr + (aux1[off] - vr) * s;         // v + (v_first - v)*sigmoid
    }
    return acc;
}

template <int EPI, bool MIXA>
__global__ __launch_bounds__(128) void gemm_wmma_kernel(
    const float* __restrict__ A, const float* __restrict__ Axx,
    const float* __restrict__ mix, const _Float16* __restrict__ BT,
    float* __restrict__ Cout, int K, int Nfull,
    const float* __restrict__ bias, const float* __restrict__ aux0,
    const float* __restrict__ aux1) {
    const int lane = threadIdx.x & 31;
    const int wv   = threadIdx.x >> 5;
    const int l    = lane & 15;
    const int hi   = lane >> 4;
    const int m0   = blockIdx.x * 64 + wv * 16;
    const int n0   = blockIdx.y * 32;
    const int m    = m0 + l;

    const float*    arow  = A + (size_t)m * K;
    const float*    axrow = Axx + (size_t)m * K;   // unused when !MIXA
    const _Float16* brow0 = BT + (size_t)(n0 + l) * K;
    const _Float16* brow1 = BT + (size_t)(n0 + 16 + l) * K;

    v8f acc0 = {};
    v8f acc1 = {};
    const int kh = hi * 8;

    for (int k0 = 0; k0 < K; k0 += 32) {
        v16h af, bf0, bf1;
#pragma unroll
        for (int e = 0; e < 8; ++e) {
            int ka = k0 + kh + e;            // K = 0..7 / 8..15 half
            int kb = k0 + 16 + kh + e;       // K = 16..23 / 24..31 half
            float xa = arow[ka], xb = arow[kb];
            if (MIXA) {
                xa += axrow[ka] * mix[ka];
                xb += axrow[kb] * mix[kb];
            }
            af[e]     = (_Float16)xa;
            af[8 + e] = (_Float16)xb;
            bf0[e]     = brow0[ka];
            bf0[8 + e] = brow0[kb];
            bf1[e]     = brow1[ka];
            bf1[8 + e] = brow1[kb];
        }
        acc0 = __builtin_amdgcn_wmma_f32_16x16x32_f16(false, af, false, bf0,
                                                      (short)0, acc0, false, false);
        acc1 = __builtin_amdgcn_wmma_f32_16x16x32_f16(false, af, false, bf1,
                                                      (short)0, acc1, false, false);
    }

    // D layout: element r -> row m0 + hi*8 + r, col n0 (+16) + l
#pragma unroll
    for (int rr = 0; rr < 8; ++rr) {
        int mm = m0 + hi * 8 + rr;
        {
            int nn   = n0 + l;
            size_t o = (size_t)mm * Nfull + nn;
            Cout[o]  = epilogue_<EPI>(acc0[rr], nn, o, bias, aux0, aux1);
        }
        {
            int nn   = n0 + 16 + l;
            size_t o = (size_t)mm * Nfull + nn;
            Cout[o]  = epilogue_<EPI>(acc1[rr], nn, o, bias, aux0, aux1);
        }
    }
}

// ---------------------------------------------------------------------------
// 64-thread (2-wave) block reduction helper.
// ---------------------------------------------------------------------------
__device__ __forceinline__ float block64_sum(float v, float* lds) {
    int tid = threadIdx.x;
    lds[tid] = v;
    __syncthreads();
    if (tid < 32) {
        float s = lds[tid] + lds[tid + 32];
#pragma unroll
        for (int off = 16; off > 0; off >>= 1) s += __shfl_down(s, off, 32);
        if (tid == 0) lds[0] = s;
    }
    __syncthreads();
    float r = lds[0];
    __syncthreads();
    return r;
}

// ---------------------------------------------------------------------------
// kk = normalize(k * k_k) per (b,t,h); k <- k2 = k*(1+(a-1)*k_a) in place.
// grid = B*T*H blocks of 64 threads.
// ---------------------------------------------------------------------------
__global__ __launch_bounds__(64) void kk_k2_kernel(float* __restrict__ k,
                                                   const float* __restrict__ a,
                                                   const float* __restrict__ k_k,
                                                   const float* __restrict__ k_a,
                                                   float* __restrict__ kk) {
    __shared__ float lds[64];
    int row = blockIdx.x;                 // (b*T+t)*H + h
    int h   = row & (H_ - 1);
    int bt  = row >> 3;
    int n   = threadIdx.x;
    int c   = h * 64 + n;
    size_t idx = (size_t)bt * C_ + c;
    float kv  = k[idx];
    float kkv = kv * k_k[c];
    float ss  = block64_sum(kkv * kkv, lds);
    kk[idx]   = kkv / fmaxf(sqrtf(ss), 1e-12f);
    float av  = a[idx];
    k[idx]    = kv * (1.f + (av - 1.f) * k_a[c]);
}

// ---------------------------------------------------------------------------
// Bidirectional RWKV7 scan.  One block per (dir*8+b, h): 256 threads,
// thread (i = tid>>2, q = tid&3) owns S[i][16q .. 16q+15].
// y2 layout: [dir][b][t][c], backward output pre-flipped to original t.
// ---------------------------------------------------------------------------
__global__ __launch_bounds__(256) void rwkv_scan_kernel(
    const float* __restrict__ r, const float* __restrict__ dec,
    const float* __restrict__ k2, const float* __restrict__ v,
    const float* __restrict__ kk, const float* __restrict__ a,
    float* __restrict__ y2) {
    __shared__ float sr[64], sdec[64], sk2[64], sv[64], saa[64], sbb[64];
    __shared__ float psa[256], py[256];
    const int h   = blockIdx.x;
    const int bd  = blockIdx.y;
    const int b   = bd & 7;
    const int dir = bd >> 3;
    const int tid = threadIdx.x;
    const int i   = tid >> 2;
    const int q   = tid & 3;
    const int jb  = q * 16;

    float S[16];
#pragma unroll
    for (int jj = 0; jj < 16; ++jj) S[jj] = 0.f;

    float* yout = y2 + (size_t)dir * ((size_t)M_ * C_);

    for (int step = 0; step < T_; ++step) {
        int t = dir ? (T_ - 1 - step) : step;
        size_t base = ((size_t)(b * T_ + t)) * C_ + h * 64;
        if (tid < 64) {
            int j    = tid;
            sr[j]    = r[base + j];
            sdec[j]  = dec[base + j];
            sk2[j]   = k2[base + j];
            sv[j]    = v[base + j];
            float kv = kk[base + j];
            saa[j]   = -kv;                 // a_ = -kk
            sbb[j]   = kv * a[base + j];    // b_ = kk * a
        }
        __syncthreads();

        float p = 0.f;
#pragma unroll
        for (int jj = 0; jj < 16; ++jj) p += S[jj] * saa[jb + jj];
        psa[tid] = p;
        __syncthreads();

        float sa = psa[i * 4 + 0] + psa[i * 4 + 1] + psa[i * 4 + 2] + psa[i * 4 + 3];
        float vi = sv[i];
        float yp = 0.f;
#pragma unroll
        for (int jj = 0; jj < 16; ++jj) {
            int j = jb + jj;
            float s = S[jj] * sdec[j] + sa * sbb[j] + vi * sk2[j];
            S[jj]   = s;
            yp     += s * sr[j];
        }
        py[tid] = yp;
        __syncthreads();

        if (q == 0) {
            yout[base + i] = py[tid] + py[tid + 1] + py[tid + 2] + py[tid + 3];
        }
    }
}

// ---------------------------------------------------------------------------
// gate blend + group-norm + r_k correction + multiply by g -> z.
// grid = B*T*H blocks of 64 threads.
// ---------------------------------------------------------------------------
__global__ __launch_bounds__(64) void gate_gn_kernel(
    const float* __restrict__ xx, const float* __restrict__ gate_w,
    const float* __restrict__ y2, const float* __restrict__ gn_w,
    const float* __restrict__ gn_b, const float* __restrict__ r,
    const float* __restrict__ k2, const float* __restrict__ r_k,
    const float* __restrict__ v, const float* __restrict__ g,
    float* __restrict__ z) {
    __shared__ float lds[64];
    int row = blockIdx.x;               // (b*T+t)*H + h
    int h   = row & 7;
    int bt  = row >> 3;
    int n   = threadIdx.x;
    int c   = h * 64 + n;
    size_t idx = (size_t)bt * C_ + c;

    float gd   = block64_sum(xx[idx] * gate_w[c], lds);
    float gate = sigmoidf_(gd);
    float yf   = y2[idx];
    float yb   = y2[(size_t)M_ * C_ + idx];
    float y    = gate * yf + (1.f - gate) * yb;

    float mu  = block64_sum(y, lds) * (1.f / 64.f);
    float d   = y - mu;
    float var = block64_sum(d * d, lds) * (1.f / 64.f);
    float yn  = d * rsqrtf(var + GN_EPS_) * gn_w[c] + gn_b[c];

    float rk  = block64_sum(r[idx] * k2[idx] * r_k[c], lds);
    yn += rk * v[idx];
    z[idx] = yn * g[idx];
}

// ---------------------------------------------------------------------------
// Host launcher
// ---------------------------------------------------------------------------
extern "C" void kernel_launch(void* const* d_in, const int* in_sizes, int n_in,
                              void* d_out, int out_size, void* d_ws, size_t ws_size,
                              hipStream_t stream) {
    const float* x       = (const float*)d_in[0];
    const float* v_first = (const float*)d_in[1];
    const float* ts_w    = (const float*)d_in[2];
    const float* x_r     = (const float*)d_in[3];
    const float* x_w     = (const float*)d_in[4];
    const float* x_k     = (const float*)d_in[5];
    const float* x_v     = (const float*)d_in[6];
    const float* x_a     = (const float*)d_in[7];
    const float* x_g     = (const float*)d_in[8];
    const float* w0      = (const float*)d_in[9];
    const float* w1      = (const float*)d_in[10];
    const float* w2      = (const float*)d_in[11];
    const float* a0      = (const float*)d_in[12];
    const float* a1      = (const float*)d_in[13];
    const float* a2      = (const float*)d_in[14];
    const float* v0      = (const float*)d_in[15];
    const float* v1      = (const float*)d_in[16];
    const float* v2      = (const float*)d_in[17];
    const float* g1      = (const float*)d_in[18];
    const float* g2      = (const float*)d_in[19];
    const float* k_k     = (const float*)d_in[20];
    const float* k_a     = (const float*)d_in[21];
    const float* r_k     = (const float*)d_in[22];
    const float* gate_w  = (const float*)d_in[23];
    const float* Wr      = (const float*)d_in[24];
    const float* Wk      = (const float*)d_in[25];
    const float* Wv      = (const float*)d_in[26];
    const float* Wo      = (const float*)d_in[27];
    const float* gn_w    = (const float*)d_in[28];
    const float* gn_b    = (const float*)d_in[29];
    float* out = (float*)d_out;

    const size_t MC = (size_t)M_ * C_;  // 4,194,304 elements

    // -------- workspace carve (256B aligned) --------
    char* p = (char*)d_ws;
    auto alloc = [&](size_t bytes) -> char* {
        char* q = p;
        p += (bytes + 255) & ~(size_t)255;
        return q;
    };
    float* xxb  = (float*)alloc(MC * 4);
    float* rb   = (float*)alloc(MC * 4);
    float* kb   = (float*)alloc(MC * 4);   // becomes k2 in place
    float* vb   = (float*)alloc(MC * 4);   // becomes final v in place
    float* decb = (float*)alloc(MC * 4);
    float* ab   = (float*)alloc(MC * 4);
    float* gb   = (float*)alloc(MC * 4);
    float* kkb  = (float*)alloc(MC * 4);
    float* zb   = (float*)alloc(MC * 4);
    float* y2b  = (float*)alloc(2 * MC * 4);
    float* hw   = (float*)alloc((size_t)M_ * DLORA_ * 4);
    float* ha   = (float*)alloc((size_t)M_ * DLORA_ * 4);
    float* hv   = (float*)alloc((size_t)M_ * DLORA_ * 4);
    float* hg   = (float*)alloc((size_t)M_ * DGATE_ * 4);
    _Float16* WrT = (_Float16*)alloc((size_t)C_ * C_ * 2);
    _Float16* WkT = (_Float16*)alloc((size_t)C_ * C_ * 2);
    _Float16* WvT = (_Float16*)alloc((size_t)C_ * C_ * 2);
    _Float16* WoT = (_Float16*)alloc((size_t)C_ * C_ * 2);
    _Float16* w1T = (_Float16*)alloc((size_t)DLORA_ * C_ * 2);
    _Float16* a1T = (_Float16*)alloc((size_t)DLORA_ * C_ * 2);
    _Float16* v1T = (_Float16*)alloc((size_t)DLORA_ * C_ * 2);
    _Float16* w2T = (_Float16*)alloc((size_t)C_ * DLORA_ * 2);
    _Float16* a2T = (_Float16*)alloc((size_t)C_ * DLORA_ * 2);
    _Float16* v2T = (_Float16*)alloc((size_t)C_ * DLORA_ * 2);
    _Float16* g1T = (_Float16*)alloc((size_t)DGATE_ * C_ * 2);
    _Float16* g2T = (_Float16*)alloc((size_t)C_ * DGATE_ * 2);

    // -------- 1) depthwise conv -> xx --------
    conv_xx_kernel<<<(B_ * T_ * C_) / 256, 256, 0, stream>>>(x, ts_w, xxb);

    // -------- 2) pack weights to f16 --------
    const int CC = C_ * C_;
    cvt_f16_kernel<<<(CC + 255) / 256, 256, 0, stream>>>(Wr, WrT, CC);
    cvt_f16_kernel<<<(CC + 255) / 256, 256, 0, stream>>>(Wk, WkT, CC);
    cvt_f16_kernel<<<(CC + 255) / 256, 256, 0, stream>>>(Wv, WvT, CC);
    cvt_f16_kernel<<<(CC + 255) / 256, 256, 0, stream>>>(Wo, WoT, CC);
    transpose_f16_kernel<<<(C_ * DLORA_ + 255) / 256, 256, 0, stream>>>(w1, w1T, C_, DLORA_);
    transpose_f16_kernel<<<(C_ * DLORA_ + 255) / 256, 256, 0, stream>>>(a1, a1T, C_, DLORA_);
    transpose_f16_kernel<<<(C_ * DLORA_ + 255) / 256, 256, 0, stream>>>(v1, v1T, C_, DLORA_);
    transpose_f16_kernel<<<(DLORA_ * C_ + 255) / 256, 256, 0, stream>>>(w2, w2T, DLORA_, C_);
    transpose_f16_kernel<<<(DLORA_ * C_ + 255) / 256, 256, 0, stream>>>(a2, a2T, DLORA_, C_);
    transpose_f16_kernel<<<(DLORA_ * C_ + 255) / 256, 256, 0, stream>>>(v2, v2T, DLORA_, C_);
    transpose_f16_kernel<<<(C_ * DGATE_ + 255) / 256, 256, 0, stream>>>(g1, g1T, C_, DGATE_);
    transpose_f16_kernel<<<(DGATE_ * C_ + 255) / 256, 256, 0, stream>>>(g2, g2T, DGATE_, C_);

    // -------- 3) big GEMMs with fused token-shift mix --------
    dim3 blk(128);
    dim3 gNC(M_ / 64, C_ / 32);      // N=512
    dim3 gN32(M_ / 64, 1);           // N=32
    dim3 gN96(M_ / 64, 3);           // N=96
    gemm_wmma_kernel<EPI_NONE, true><<<gNC, blk, 0, stream>>>(
        x, xxb, x_r, WrT, rb, C_, C_, nullptr, nullptr, nullptr);
    gemm_wmma_kernel<EPI_NONE, true><<<gNC, blk, 0, stream>>>(
        x, xxb, x_k, WkT, kb, C_, C_, nullptr, nullptr, nullptr);
    gemm_wmma_kernel<EPI_NONE, true><<<gNC, blk, 0, stream>>>(
        x, xxb, x_v, WvT, vb, C_, C_, nullptr, nullptr, nullptr);

    // -------- 4) LoRA stage 1 --------
    gemm_wmma_kernel<EPI_TANH, true><<<gN32, blk, 0, stream>>>(
        x, xxb, x_w, w1T, hw, C_, DLORA_, nullptr, nullptr, nullptr);
    gemm_wmma_kernel<EPI_NONE, true><<<gN32, blk, 0, stream>>>(
        x, xxb, x_a, a1T, ha, C_, DLORA_, nullptr, nullptr, nullptr);
    gemm_wmma_kernel<EPI_NONE, true><<<gN32, blk, 0, stream>>>(
        x, xxb, x_v, v1T, hv, C_, DLORA_, nullptr, nullptr, nullptr);
    gemm_wmma_kernel<EPI_SIG, true><<<gN96, blk, 0, stream>>>(
        x, xxb, x_g, g1T, hg, C_, DGATE_, nullptr, nullptr, nullptr);

    // -------- 5) LoRA stage 2 (fused epilogues) --------
    gemm_wmma_kernel<EPI_DEC, false><<<gNC, blk, 0, stream>>>(
        hw, nullptr, nullptr, w2T, decb, DLORA_, C_, w0, nullptr, nullptr);
    gemm_wmma_kernel<EPI_SIGB, false><<<gNC, blk, 0, stream>>>(
        ha, nullptr, nullptr, a2T, ab, DLORA_, C_, a0, nullptr, nullptr);
    gemm_wmma_kernel<EPI_VMIX, false><<<gNC, blk, 0, stream>>>(
        hv, nullptr, nullptr, v2T, vb, DLORA_, C_, v0, vb, v_first);
    gemm_wmma_kernel<EPI_NONE, false><<<gNC, blk, 0, stream>>>(
        hg, nullptr, nullptr, g2T, gb, DGATE_, C_, nullptr, nullptr, nullptr);

    // -------- 6) kk + k2 --------
    kk_k2_kernel<<<M_ * H_, 64, 0, stream>>>(kb, ab, k_k, k_a, kkb);

    // -------- 7) bidirectional RWKV7 scan --------
    rwkv_scan_kernel<<<dim3(H_, 2 * B_), 256, 0, stream>>>(rb, decb, kb, vb, kkb, ab, y2b);

    // -------- 8) gate + groupnorm + r_k correction --------
    gate_gn_kernel<<<M_ * H_, 64, 0, stream>>>(xxb, gate_w, y2b, gn_w, gn_b,
                                               rb, kb, r_k, vb, gb, zb);

    // -------- 9) output projection --------
    gemm_wmma_kernel<EPI_NONE, false><<<gNC, blk, 0, stream>>>(
        zb, nullptr, nullptr, WoT, out, C_, C_, nullptr, nullptr, nullptr);

    // -------- 10) second tuple output: v_first passthrough --------
    hipMemcpyAsync(out + MC, v_first, MC * sizeof(float),
                   hipMemcpyDeviceToDevice, stream);
}
